// DeltaTDNet_23648089932155
// MI455X (gfx1250) — matrix operand order repaired and checked by reading, not compiled, not run on hardware
//
#include <hip/hip_runtime.h>
#include <hip/hip_bf16.h>

typedef __attribute__((ext_vector_type(16))) _Float16 v16h;
typedef __attribute__((ext_vector_type(8)))  _Float16 v8h;
typedef __attribute__((ext_vector_type(4)))  _Float16 v4h;
typedef __attribute__((ext_vector_type(8)))  float    v8f;

#define SNN_ALPHA 0.95f
#define SNN_BETA  0.9f
#define SNN_B 256
#define SNN_D 1024
#define SNN_H 2048
#define SNN_T 100

// ---------------- WMMA fragment loaders (wave32, 16x16x32 f16) ----------------
// A (16x32, MxK), row-major source [.., ldk]:
//   lanes 0-15: row M=lane,   K = k0+{0..7} (v0-3), k0+{16..23} (v4-7)
//   lanes16-31: row M=lane-16,K = k0+{8..15}(v0-3), k0+{24..31}(v4-7)
static __device__ __forceinline__ v16h load_a_frag(const _Float16* __restrict__ base,
                                                   int row0, int ldk, int k0, int lane) {
  const int row = row0 + (lane & 15);
  const _Float16* p = base + (size_t)row * ldk + (k0 + ((lane >> 4) << 3));
  union { v16h v; v8h h[2]; } u;
  u.h[0] = *(const v8h*)(p);
  u.h[1] = *(const v8h*)(p + 16);
  return u.v;
}
// B (32x16, KxN) from a pre-transposed weight wt[N][K] (row n = column n of W):
//   lanes 0-15: col N=lane,    K = k0+{0..7} (v0-3), k0+{16..23} (v4-7)
//   lanes16-31: col N=lane-16, K = k0+{8..15}(v0-3), k0+{24..31}(v4-7)
static __device__ __forceinline__ v16h load_b_frag(const _Float16* __restrict__ wt,
                                                   int col0, int ldk, int k0, int lane) {
  const int col = col0 + (lane & 15);
  const _Float16* p = wt + (size_t)col * ldk + (k0 + ((lane >> 4) << 3));
  union { v16h v; v8h h[2]; } u;
  u.h[0] = *(const v8h*)(p);
  u.h[1] = *(const v8h*)(p + 16);
  return u.v;
}

// ---------------- one-time operand preparation ----------------
__global__ void split_plain(const float* __restrict__ in,
                            _Float16* __restrict__ hi, _Float16* __restrict__ lo, int n) {
  int i = blockIdx.x * blockDim.x + threadIdx.x;
  if (i < n) {
    float v = in[i];
    _Float16 h = (_Float16)v;
    hi[i] = h;
    lo[i] = (_Float16)(v - (float)h);
  }
}
// in: [K][N] row-major -> out hi/lo: [N][K] (column-major of in), f32 = hi + lo
__global__ void split_transpose(const float* __restrict__ in,
                                _Float16* __restrict__ hi, _Float16* __restrict__ lo,
                                int K, int N) {
  long i = (long)blockIdx.x * blockDim.x + threadIdx.x;
  if (i < (long)N * K) {
    int n = (int)(i / K);
    int k = (int)(i - (long)n * K);
    float v = in[(size_t)k * N + n];
    _Float16 h = (_Float16)v;
    hi[i] = h;
    lo[i] = (_Float16)(v - (float)h);
  }
}

// ---------------- one-time GEMM: xw0 = x @ w0 (f32-accurate via hi/lo split) ----------------
// grid: 64 blocks x 256 thr; block tile 128M x 64N; wave tile 64M x 16N
__global__ __launch_bounds__(256) void xw0_gemm(
    const _Float16* __restrict__ x_hi, const _Float16* __restrict__ x_lo,     // [B][D]
    const _Float16* __restrict__ w0t_hi, const _Float16* __restrict__ w0t_lo, // [H][D]
    float* __restrict__ xw0)                                                  // [B][H]
{
  const int lane = threadIdx.x & 31;
  const int w    = threadIdx.x >> 5;
  const int mbase = (blockIdx.x & 1) * 128 + (w & 1) * 64;
  const int ncol  = (blockIdx.x >> 1) * 64 + (w >> 1) * 16;

  v8f acc[4];
#pragma unroll
  for (int i = 0; i < 4; ++i)
#pragma unroll
    for (int e = 0; e < 8; ++e) acc[i][e] = 0.0f;

  for (int k0 = 0; k0 < SNN_D; k0 += 32) {
    v16h ah[4], al[4];
#pragma unroll
    for (int mt = 0; mt < 4; ++mt) {
      ah[mt] = load_a_frag(x_hi, mbase + mt * 16, SNN_D, k0, lane);
      al[mt] = load_a_frag(x_lo, mbase + mt * 16, SNN_D, k0, lane);
    }
    v16h bh = load_b_frag(w0t_hi, ncol, SNN_D, k0, lane);
    v16h bl = load_b_frag(w0t_lo, ncol, SNN_D, k0, lane);
#pragma unroll
    for (int mt = 0; mt < 4; ++mt) {
      acc[mt] = __builtin_amdgcn_wmma_f32_16x16x32_f16(false, ah[mt], false, bh, (short)0, acc[mt], false, false);
      acc[mt] = __builtin_amdgcn_wmma_f32_16x16x32_f16(false, ah[mt], false, bl, (short)0, acc[mt], false, false);
      acc[mt] = __builtin_amdgcn_wmma_f32_16x16x32_f16(false, al[mt], false, bh, (short)0, acc[mt], false, false);
    }
  }
  // C/D layout: lane 0-15 -> col=lane, vgpr r -> row r; lanes 16-31 -> row r+8
#pragma unroll
  for (int mt = 0; mt < 4; ++mt)
#pragma unroll
    for (int r = 0; r < 8; ++r) {
      int row = mbase + mt * 16 + r + ((lane >> 4) << 3);
      int col = ncol + (lane & 15);
      xw0[(size_t)row * SNN_H + col] = acc[mt][r];
    }
}

// ---------------- per-step elementwise LIF layer-0 update + spike ----------------
__global__ __launch_bounds__(256) void spike_step(
    const float4* __restrict__ xw0, const float4* __restrict__ m0_prev,
    float4* __restrict__ m0_out, float4* __restrict__ spk_out,
    v4h* __restrict__ z16)
{
  int i = blockIdx.x * blockDim.x + threadIdx.x;   // exactly B*H/4 threads
  float4 m = m0_prev[i];
  float4 c = xw0[i];
  float nm[4] = { SNN_BETA * m.x + c.x, SNN_BETA * m.y + c.y,
                  SNN_BETA * m.z + c.z, SNN_BETA * m.w + c.w };
  float4 zf; v4h zh;
#pragma unroll
  for (int e = 0; e < 4; ++e) {
    float mthr = nm[e] - 1.0f;
    bool sp = mthr > 0.0f;
    float z = sp ? 1.0f : 0.0f;
    nm[e] = sp ? 0.0f : nm[e];
    ((float*)&zf)[e] = z;
    zh[e] = (_Float16)z;
  }
  m0_out[i]  = make_float4(nm[0], nm[1], nm[2], nm[3]);
  spk_out[i] = zf;
  z16[i]     = zh;
}

// ---------------- per-step GEMM: ns1 = ALPHA*s1 + z @ w1 ; nm1 = BETA*m1 + ns1 ----------------
__global__ __launch_bounds__(256) void step_gemm(
    const _Float16* __restrict__ z16,                                        // [B][H]
    const _Float16* __restrict__ w1t_hi, const _Float16* __restrict__ w1t_lo,// [H][H]
    float* __restrict__ s1,                                                  // [B][H] state
    const float* __restrict__ m1_prev,                                       // mem_rec[t-1,1]
    float* __restrict__ m1_out,                                              // mem_rec[t,1]
    float* __restrict__ out0)                                                // final membrane or null
{
  const int lane = threadIdx.x & 31;
  const int w    = threadIdx.x >> 5;
  const int mbase = (blockIdx.x & 1) * 128 + (w & 1) * 64;
  const int ncol  = (blockIdx.x >> 1) * 64 + (w >> 1) * 16;

  v8f acc[4];
#pragma unroll
  for (int i = 0; i < 4; ++i)
#pragma unroll
    for (int e = 0; e < 8; ++e) acc[i][e] = 0.0f;

  for (int k0 = 0; k0 < SNN_H; k0 += 32) {
    v16h a[4];
#pragma unroll
    for (int mt = 0; mt < 4; ++mt)
      a[mt] = load_a_frag(z16, mbase + mt * 16, SNN_H, k0, lane);
    v16h bh = load_b_frag(w1t_hi, ncol, SNN_H, k0, lane);
    v16h bl = load_b_frag(w1t_lo, ncol, SNN_H, k0, lane);
#pragma unroll
    for (int mt = 0; mt < 4; ++mt) {
      acc[mt] = __builtin_amdgcn_wmma_f32_16x16x32_f16(false, a[mt], false, bh, (short)0, acc[mt], false, false);
      acc[mt] = __builtin_amdgcn_wmma_f32_16x16x32_f16(false, a[mt], false, bl, (short)0, acc[mt], false, false);
    }
  }
#pragma unroll
  for (int mt = 0; mt < 4; ++mt)
#pragma unroll
    for (int r = 0; r < 8; ++r) {
      int row = mbase + mt * 16 + r + ((lane >> 4) << 3);
      int col = ncol + (lane & 15);
      size_t idx = (size_t)row * SNN_H + col;
      float ns1 = SNN_ALPHA * s1[idx] + acc[mt][r];
      float nm1 = SNN_BETA * m1_prev[idx] + ns1;
      s1[idx] = ns1;
      m1_out[idx] = nm1;
      if (out0) out0[idx] = nm1;
    }
}

extern "C" void kernel_launch(void* const* d_in, const int* in_sizes, int n_in,
                              void* d_out, int out_size, void* d_ws, size_t ws_size,
                              hipStream_t stream) {
  const float* x  = (const float*)d_in[0];  // [256,1024]
  const float* w0 = (const float*)d_in[1];  // [1024,2048]
  const float* w1 = (const float*)d_in[2];  // [2048,2048]

  const size_t BH = (size_t)SNN_B * SNN_H;
  float* out0    = (float*)d_out;                         // mem1_rec[-1]  [B,H]
  float* mem_rec = out0 + BH;                             // [T,2,B,H]
  float* spk_rec = mem_rec + (size_t)SNN_T * 2 * BH;      // [T,B,H]

  // workspace carve-up (256B aligned)
  char* ws = (char*)d_ws;
  size_t off = 0;
  auto carve = [&](size_t bytes) { void* p = ws + off; off += (bytes + 255) & ~(size_t)255; return p; };
  float*    xw0    = (float*)   carve(BH * 4);
  float*    s1     = (float*)   carve(BH * 4);
  _Float16* z16    = (_Float16*)carve(BH * 2);
  _Float16* x_hi   = (_Float16*)carve((size_t)SNN_B * SNN_D * 2);
  _Float16* x_lo   = (_Float16*)carve((size_t)SNN_B * SNN_D * 2);
  _Float16* w0t_hi = (_Float16*)carve((size_t)SNN_H * SNN_D * 2);
  _Float16* w0t_lo = (_Float16*)carve((size_t)SNN_H * SNN_D * 2);
  _Float16* w1t_hi = (_Float16*)carve((size_t)SNN_H * SNN_H * 2);
  _Float16* w1t_lo = (_Float16*)carve((size_t)SNN_H * SNN_H * 2);

  // zero t=0 records (they ARE the initial state read at t=1) and s1 state
  hipMemsetAsync(mem_rec, 0, 2 * BH * 4, stream);
  hipMemsetAsync(spk_rec, 0, BH * 4, stream);
  hipMemsetAsync(s1, 0, BH * 4, stream);

  // one-time operand prep
  {
    int n = SNN_B * SNN_D;
    split_plain<<<(n + 255) / 256, 256, 0, stream>>>(x, x_hi, x_lo, n);
  }
  split_transpose<<<(SNN_D * SNN_H + 255) / 256, 256, 0, stream>>>(w0, w0t_hi, w0t_lo, SNN_D, SNN_H);
  split_transpose<<<(SNN_H * SNN_H + 255) / 256, 256, 0, stream>>>(w1, w1t_hi, w1t_lo, SNN_H, SNN_H);

  // loop-invariant layer-0 drive: xw0 = x @ w0 (computed once, reused 99x)
  xw0_gemm<<<64, 256, 0, stream>>>(x_hi, x_lo, w0t_hi, w0t_lo, xw0);

  // 99 recurrent steps; stream order provides the z_t -> GEMM dependency
  for (int t = 1; t < SNN_T; ++t) {
    const float* m0_prev = mem_rec + ((size_t)(t - 1) * 2 + 0) * BH;
    float* m0_out  = mem_rec + ((size_t)t * 2 + 0) * BH;
    float* spk_out = spk_rec + (size_t)t * BH;
    spike_step<<<(int)(BH / 4 / 256), 256, 0, stream>>>(
        (const float4*)xw0, (const float4*)m0_prev,
        (float4*)m0_out, (float4*)spk_out, (v4h*)z16);

    const float* m1_prev = mem_rec + ((size_t)(t - 1) * 2 + 1) * BH;
    float* m1_out = mem_rec + ((size_t)t * 2 + 1) * BH;
    step_gemm<<<64, 256, 0, stream>>>(z16, w1t_hi, w1t_lo, s1, m1_prev, m1_out,
                                      (t == SNN_T - 1) ? out0 : nullptr);
  }
}
